// MixerBlock_27230092656751
// MI455X (gfx1250) — matrix operand order repaired and checked by reading, not compiled, not run on hardware
//
#include <hip/hip_runtime.h>
#include <hip/hip_bf16.h>

// ---------------------------------------------------------------------------
// MixerBlock for MI455X (gfx1250, wave32, WMMA bf16 16x16x32, f32 accum)
// B=4, S=4096, D=1024, H=8, E=128, EXP=4, DC=8
// ---------------------------------------------------------------------------

typedef __attribute__((ext_vector_type(16))) __bf16 v16bf;
typedef __attribute__((ext_vector_type(8)))  float  v8f;

#define BB 4
#define SS 4096
#define DD 1024
#define HH 8
#define EE 128
#define MM (BB * SS)          // 16384 tokens
#define FF (4 * DD)           // 4096 MLP hidden

// ------------------------------- helpers -----------------------------------

__global__ __launch_bounds__(256) void cast_f32_to_bf16(
    const float* __restrict__ in, __bf16* __restrict__ out, int n) {
  int i = blockIdx.x * 256 + threadIdx.x;
  if (i < n) out[i] = (__bf16)in[i];
}

// LayerNorm over D=1024, one block per token row, output bf16.
__global__ __launch_bounds__(256) void ln_to_bf16(
    const float* __restrict__ x, const float* __restrict__ g,
    const float* __restrict__ b, __bf16* __restrict__ out) {
  const int row = blockIdx.x;
  const float* xr = x + (size_t)row * DD;
  __shared__ float red0[8], red1[8];
  float s = 0.f, s2 = 0.f;
  for (int i = threadIdx.x; i < DD; i += 256) {
    float v = xr[i];
    s += v; s2 += v * v;
  }
  for (int off = 16; off > 0; off >>= 1) {
    s  += __shfl_down(s,  off, 32);
    s2 += __shfl_down(s2, off, 32);
  }
  const int wave = threadIdx.x >> 5, lane = threadIdx.x & 31;
  if (lane == 0) { red0[wave] = s; red1[wave] = s2; }
  __syncthreads();
  float ts = 0.f, ts2 = 0.f;
  for (int w = 0; w < 8; ++w) { ts += red0[w]; ts2 += red1[w]; }
  const float mean = ts * (1.0f / DD);
  const float var  = ts2 * (1.0f / DD) - mean * mean;
  const float inv  = rsqrtf(var + 1e-5f);
  for (int i = threadIdx.x; i < DD; i += 256) {
    float v = (xr[i] - mean) * inv * g[i] + b[i];
    out[(size_t)row * DD + i] = (__bf16)v;
  }
}

// ------------------------------ WMMA GEMM ----------------------------------
// C[M,N] = A[M,K] * Bw[N,K]^T (+ epilogue).  A, Bw bf16 row-major.
// EPI 0: silu(acc+bias) -> bf16 Cb            (MLP up)
// EPI 1: acc+bias+resid -> f32 Cf             (MLP down + residual x)
// EPI 2: (acc+bias)*c1(head,s) -> f32 Cf      (head proj; row-head pre-scale)
// EPI 3: acc+bias+resid -> f32 Cf             (out proj + residual x2)
template <int EPI>
__global__ __launch_bounds__(256) void gemm_bf16(
    const __bf16* __restrict__ A, const __bf16* __restrict__ Bw,
    const float* __restrict__ bias, const float* __restrict__ resid,
    const float* __restrict__ mixw,
    float* __restrict__ Cf, __bf16* __restrict__ Cb,
    int N, int K) {
  const int lane = threadIdx.x & 31;
  const int wave = threadIdx.x >> 5;
  const int wm = wave & 3;                // 4 wave-rows
  const int wn = wave >> 2;               // 2 wave-cols
  const int m_wave = blockIdx.y * 128 + wm * 32;
  const int n_wave = blockIdx.x * 128 + wn * 64;

  const int arow = lane & 15;
  const int ksel = (lane >> 4) & 1;       // which K half this lane holds

  v8f acc[2][4] = {};

  for (int k0 = 0; k0 < K; k0 += 32) {
    // A fragments: 16x32, lane<16 -> K {0..7,16..23}, lane>=16 -> {8..15,24..31}
    v16bf afr[2];
#pragma unroll
    for (int i = 0; i < 2; ++i) {
      const __bf16* ap =
          A + (size_t)(m_wave + i * 16 + arow) * K + k0 + ksel * 8;
      uint4* dst = (uint4*)&afr[i];
      dst[0] = *(const uint4*)ap;          // K = base .. base+7
      dst[1] = *(const uint4*)(ap + 16);   // K = base+16 .. base+23
    }
    // B fragments: 32x16, lane<16 -> K {0..7,16..23}, lane>=16 -> {8..15,24..31}
    v16bf bfr[4];
#pragma unroll
    for (int j = 0; j < 4; ++j) {
      const __bf16* bp =
          Bw + (size_t)(n_wave + j * 16 + arow) * K + k0 + ksel * 8;
      uint4* dst = (uint4*)&bfr[j];
      dst[0] = *(const uint4*)bp;
      dst[1] = *(const uint4*)(bp + 16);
    }
#pragma unroll
    for (int i = 0; i < 2; ++i)
#pragma unroll
      for (int j = 0; j < 4; ++j)
        acc[i][j] = __builtin_amdgcn_wmma_f32_16x16x32_bf16(
            false, afr[i], false, bfr[j], (short)0, acc[i][j], false, false);
  }

  // Epilogue.  C/D layout: VGPR r -> row m_wave+i*16+r+(lane>=16?8:0),
  // col n_wave+j*16+(lane&15).
#pragma unroll
  for (int i = 0; i < 2; ++i) {
#pragma unroll
    for (int j = 0; j < 4; ++j) {
      const int col = n_wave + j * 16 + (lane & 15);
#pragma unroll
      for (int r = 0; r < 8; ++r) {
        const int row = m_wave + i * 16 + r + (ksel << 3);
        float v = acc[i][j][r] + bias[col];
        const size_t idx = (size_t)row * N + col;
        if constexpr (EPI == 0) {
          v = v / (1.0f + __expf(-v));     // SiLU
          Cb[idx] = (__bf16)v;
        } else if constexpr (EPI == 1) {
          Cf[idx] = v + resid[idx];
        } else if constexpr (EPI == 2) {
          const int h = col >> 7;          // E = 128
          if (h >= HH / 2) v *= mixw[h * SS + (row & (SS - 1))];  // Row heads
          Cf[idx] = v;
        } else {
          Cf[idx] = v + resid[idx];
        }
      }
    }
  }
}

// ------------------------- chunked decayed scan ----------------------------
// cache_t = u_t + d*cache_{t-1};  32 chunks of 128 steps, 131072-way parallel.

__device__ __forceinline__ float head_decay(const float* dr, int h) {
  float d = fminf(fmaxf(dr[h], 0.9f), 1.0f);
  return powf(d, 0.125f);                  // ^(1/DC), DC = 8
}

__global__ __launch_bounds__(256) void scan_partial(
    const float* __restrict__ u, const float* __restrict__ decay_raw,
    float* __restrict__ chunk_end) {
  const int tid = blockIdx.x * 256 + threadIdx.x;   // 131072
  const int e = tid & 127;
  const int c = (tid >> 7) & 31;
  const int hb = tid >> 12;                // h*4 + b
  const int b = hb & 3, h = hb >> 2;
  const float d = head_decay(decay_raw, h);
  const float* up = u + ((size_t)b * SS + c * 128) * DD + h * EE + e;
  float cache = 0.f;
  for (int i = 0; i < 128; ++i) cache = up[(size_t)i * DD] + d * cache;
  chunk_end[(size_t)(hb * 32 + c) * 128 + e] = cache;
}

__global__ __launch_bounds__(256) void scan_prefix(
    const float* __restrict__ chunk_end, const float* __restrict__ decay_raw,
    float* __restrict__ pref) {
  const int tid = blockIdx.x * 256 + threadIdx.x;   // 4096
  const int e = tid & 127;
  const int hb = tid >> 7;
  const int h = hb >> 2;
  const float d = head_decay(decay_raw, h);
  float dL = d;
  for (int i = 0; i < 7; ++i) dL *= dL;             // d^128
  float p = 0.f;
  for (int c = 0; c < 32; ++c) {
    const size_t idx = (size_t)(hb * 32 + c) * 128 + e;
    pref[idx] = p;
    p = chunk_end[idx] + dL * p;
  }
}

__global__ __launch_bounds__(256) void scan_final(
    const float* __restrict__ u, const float* __restrict__ pref,
    const float* __restrict__ decay_raw, const float* __restrict__ mix_w,
    const float* __restrict__ mix_b, __bf16* __restrict__ mixed) {
  const int tid = blockIdx.x * 256 + threadIdx.x;
  const int e = tid & 127;
  const int c = (tid >> 7) & 31;
  const int hb = tid >> 12;
  const int b = hb & 3, h = hb >> 2;
  const float d = head_decay(decay_raw, h);
  const bool isCol = (h < HH / 2);
  const float* up = u + ((size_t)b * SS + c * 128) * DD + h * EE + e;
  const float* mwp = mix_w + h * SS;
  const float* mbp = mix_b + h * SS;
  float cache = pref[(size_t)(hb * 32 + c) * 128 + e];
  for (int i = 0; i < 128; ++i) {
    const int s = c * 128 + i;
    cache = up[(size_t)i * DD] + d * cache;
    const float c2 = isCol ? mwp[s] : 1.0f;
    const float v = c2 * cache + mbp[s];
    mixed[((size_t)b * SS + s) * DD + h * EE + e] = (__bf16)v;
  }
}

// ------------------------------- launcher ----------------------------------

extern "C" void kernel_launch(void* const* d_in, const int* in_sizes, int n_in,
                              void* d_out, int out_size, void* d_ws,
                              size_t ws_size, hipStream_t stream) {
  const float* x        = (const float*)d_in[0];
  const float* ln1_g    = (const float*)d_in[1];
  const float* ln1_b    = (const float*)d_in[2];
  const float* w1       = (const float*)d_in[3];
  const float* b1       = (const float*)d_in[4];
  const float* w2       = (const float*)d_in[5];
  const float* b2       = (const float*)d_in[6];
  const float* ln2_g    = (const float*)d_in[7];
  const float* ln2_b    = (const float*)d_in[8];
  const float* proj_w   = (const float*)d_in[9];
  const float* proj_b   = (const float*)d_in[10];
  const float* mix_w    = (const float*)d_in[11];
  const float* mix_b    = (const float*)d_in[12];
  const float* decay_raw= (const float*)d_in[13];
  const float* out_w    = (const float*)d_in[14];
  const float* out_b    = (const float*)d_in[15];
  float* out            = (float*)d_out;

  char* ws = (char*)d_ws;
  const size_t MB = 1ull << 20;
  __bf16* w1b  = (__bf16*)(ws);               //  8 MB (4M bf16)
  __bf16* w2b  = (__bf16*)(ws + 8 * MB);      //  8 MB
  __bf16* pwb  = (__bf16*)(ws + 16 * MB);     //  2 MB
  __bf16* owb  = (__bf16*)(ws + 18 * MB);     //  2 MB
  __bf16* act  = (__bf16*)(ws + 20 * MB);     // 32 MB  (ln1, later ln2)
  char*   big  = ws + 52 * MB;                // 128 MB region
  __bf16* h1   = (__bf16*)big;                //  h1 bf16 [M,4096] (128 MB)
  float*  u    = (float*)big;                 //  u  f32  [M,1024] (64 MB, reuse)
  __bf16* mixed= (__bf16*)(big + 96 * MB);    //  mixed bf16 [M,1024] (32 MB)
  float*  x2   = (float*)(ws + 180 * MB);     // 64 MB
  float*  cend = (float*)(ws + 244 * MB);     // 512 KB
  float*  cpref= (float*)(ws + 244 * MB + 512 * 1024);

  // 1) cast weights to bf16
  cast_f32_to_bf16<<<(FF * DD + 255) / 256, 256, 0, stream>>>(w1, w1b, FF * DD);
  cast_f32_to_bf16<<<(DD * FF + 255) / 256, 256, 0, stream>>>(w2, w2b, DD * FF);
  cast_f32_to_bf16<<<(DD * DD + 255) / 256, 256, 0, stream>>>(proj_w, pwb, DD * DD);
  cast_f32_to_bf16<<<(DD * DD + 255) / 256, 256, 0, stream>>>(out_w, owb, DD * DD);

  // 2) LN1 -> bf16
  ln_to_bf16<<<MM, 256, 0, stream>>>(x, ln1_g, ln1_b, act);

  // 3) MLP up: h1 = silu(ln1 @ w1^T + b1)
  gemm_bf16<0><<<dim3(FF / 128, MM / 128), 256, 0, stream>>>(
      act, w1b, b1, nullptr, nullptr, nullptr, h1, FF, DD);

  // 4) MLP down + residual: x2 = h1 @ w2^T + b2 + x
  gemm_bf16<1><<<dim3(DD / 128, MM / 128), 256, 0, stream>>>(
      h1, w2b, b2, x, nullptr, x2, nullptr, DD, FF);

  // 5) LN2 -> bf16 (reuse act)
  ln_to_bf16<<<MM, 256, 0, stream>>>(x2, ln2_g, ln2_b, act);

  // 6) head projection (+proj_b, Row-head pre-scale c1): u (f32, reuses h1)
  gemm_bf16<2><<<dim3(DD / 128, MM / 128), 256, 0, stream>>>(
      act, pwb, proj_b, nullptr, mix_w, u, nullptr, DD, DD);

  // 7) chunked decayed scan -> mixed (bf16)
  scan_partial<<<(HH * BB * 32 * EE) / 256, 256, 0, stream>>>(u, decay_raw, cend);
  scan_prefix<<<(HH * BB * EE) / 256, 256, 0, stream>>>(cend, decay_raw, cpref);
  scan_final<<<(HH * BB * 32 * EE) / 256, 256, 0, stream>>>(
      u, cpref, decay_raw, mix_w, mix_b, mixed);

  // 8) out projection + residual: out = mixed @ out_w^T + out_b + x2
  gemm_bf16<3><<<dim3(DD / 128, MM / 128), 256, 0, stream>>>(
      mixed, owb, out_b, x2, nullptr, out, nullptr, DD, DD);
}
